// LSTMAutoencoder_42932493091034
// MI455X (gfx1250) — compile-verified, matrix operand order checked by reading, not run on hardware
//
#include <hip/hip_runtime.h>
#include <hip/hip_bf16.h>

typedef __attribute__((ext_vector_type(16))) _Float16 v16h;
typedef __attribute__((ext_vector_type(8)))  float    v8f;

#define MATN   262144   // 1024*256 elements per weight matrix
#define NWG    16

__device__ __forceinline__ float sigf(float v)      { return 1.0f / (1.0f + __expf(-v)); }
__device__ __forceinline__ float tanhfast(float v)  { return 1.0f - 2.0f / (1.0f + __expf(2.0f * v)); }

// ---------------- prep: f32 -> f16 weight conversion (6 matrices) ----------------
__global__ void k_conv6(const float* __restrict__ a0, const float* __restrict__ a1,
                        const float* __restrict__ a2, const float* __restrict__ a3,
                        const float* __restrict__ a4, const float* __restrict__ a5,
                        _Float16* __restrict__ dst) {
    unsigned i = blockIdx.x * 256u + threadIdx.x;      // 6*262144 total
    unsigned m = i >> 18, r = i & (MATN - 1);
    const float* s;
    switch (m) {
        case 0: s = a0; break; case 1: s = a1; break; case 2: s = a2; break;
        case 3: s = a3; break; case 4: s = a4; break; default: s = a5; break;
    }
    dst[i] = (_Float16)s[r];
}

// ---------------- prep: fused biases, zero state, decoder seed, barrier ----------------
__global__ void k_misc(const float* b0a, const float* b0b, const float* b1a, const float* b1b,
                       const float* b2a, const float* b2b, const float* b3a, const float* b3b,
                       const float* __restrict__ x, float* __restrict__ biases,
                       _Float16* __restrict__ h0, _Float16* __restrict__ h1,
                       float* __restrict__ dec_in, unsigned* __restrict__ bar) {
    unsigned i = blockIdx.x * 256u + threadIdx.x;      // 65536 threads
    if (i < 4096u) {
        unsigned g = i >> 10, j = i & 1023u;
        const float* pa = (g == 0) ? b0a : (g == 1) ? b1a : (g == 2) ? b2a : b3a;
        const float* pb = (g == 0) ? b0b : (g == 1) ? b1b : (g == 2) ? b2b : b3b;
        biases[i] = pa[j] + pb[j];
    }
    if (i < 65536u) { h0[i] = (_Float16)0.0f; h1[i] = (_Float16)0.0f; }  // both double-buffers
    if (i < 128u)   dec_in[i] = x[i * 512u];                              // x[:,0,0]
    if (i < 2u)     bar[i] = 0u;
}

// ---------------- grid-wide barrier (persistent kernel, 16 co-resident WGs) ----------------
__device__ __forceinline__ void grid_barrier(unsigned* cnt, unsigned* gen) {
    __syncthreads();
    if (threadIdx.x == 0) {
        __threadfence();
        unsigned g = __hip_atomic_load(gen, __ATOMIC_ACQUIRE, __HIP_MEMORY_SCOPE_AGENT);
        unsigned a = __hip_atomic_fetch_add(cnt, 1u, __ATOMIC_ACQ_REL, __HIP_MEMORY_SCOPE_AGENT);
        if (a == NWG - 1u) {
            __hip_atomic_store(cnt, 0u, __ATOMIC_RELAXED, __HIP_MEMORY_SCOPE_AGENT);
            __hip_atomic_fetch_add(gen, 1u, __ATOMIC_ACQ_REL, __HIP_MEMORY_SCOPE_AGENT);
        } else {
            while (__hip_atomic_load(gen, __ATOMIC_ACQUIRE, __HIP_MEMORY_SCOPE_AGENT) == g)
                __builtin_amdgcn_s_sleep(1);
        }
    }
    __syncthreads();
}

// ---------------- one GEMM slab: acc(4 M-tiles x 1 N-tile) += A(h) x W^T ----------------
// A: f16 [128][256] row-major (h state). W: f16 [1024][256] row-major (gate rows).
// B-frag lane layout (this code's contiguous k order, shared with A): lane holds column n = ln&15, K halfs at k*32 + (ln>>4)*16,
// value B[k][n] = W[n][k] -> contiguous 32B in W row `ncol`.
__device__ __forceinline__ void gemm_slab(const _Float16* __restrict__ A,
                                          const _Float16* __restrict__ W,
                                          int m0, int arow, int kb, int ncol,
                                          v8f acc[4]) {
    const _Float16* wbase = W + ncol * 256 + kb;
    const _Float16* abase = A + (m0 + arow) * 256 + kb;
    #pragma unroll
    for (int k = 0; k < 8; ++k) {
        v16h bf = *(const v16h*)(wbase + k * 32);
        #pragma unroll
        for (int mt = 0; mt < 4; ++mt) {
            v16h af = *(const v16h*)(abase + mt * 16 * 256 + k * 32);
            acc[mt] = __builtin_amdgcn_wmma_f32_16x16x32_f16(
                false, af, false, bf, (short)0, acc[mt], false, false);
        }
    }
}

// ---------------- persistent seq2seq kernel ----------------
__global__ __launch_bounds__(256, 1) void lstm_seq2seq(
    const float* __restrict__ x,
    const float* __restrict__ eWih0, const float* __restrict__ dWih0,
    const float* __restrict__ outW,  const float* __restrict__ outB,
    const _Float16* __restrict__ w6, const float* __restrict__ biases,
    _Float16* __restrict__ h0buf, _Float16* __restrict__ h1buf,
    float* __restrict__ dec_in, unsigned* __restrict__ bar,
    float* __restrict__ out) {

    __shared__ float s_gate[4 * 64 * 32];   // [gate][64 rows][32 cols] pre-activation exchange

    const int tid = threadIdx.x;
    const int wg  = blockIdx.x;
    const int mg  = wg >> 3;                // batch group (0..1): rows m0..m0+63
    const int cg  = wg & 7;                 // hidden slice (0..7): cols cg*32..+31
    const int m0  = mg * 64;
    const int wv  = tid >> 5, ln = tid & 31;
    const int gate = wv >> 1;               // 0..3 : i,f,g,o
    const int jl0  = (wv & 1) * 16;         // N-tile offset within 32-col slice
    const int ncol = gate * 256 + cg * 32 + jl0 + (ln & 15);  // global gate column
    const int kb   = (ln >> 4) * 16;        // K offset within 32-K block
    const int arow = ln & 15;               // A row within M-tile

    // cell state: 8 (b,j) elements per thread, resident in registers for all 1024 steps
    float c0[8], c1[8];
    #pragma unroll
    for (int q = 0; q < 8; ++q) { c0[q] = 0.0f; c1[q] = 0.0f; }

    unsigned* bcnt = bar;
    unsigned* bgen = bar + 1;
    int p = 0;

    for (int phase = 0; phase < 2; ++phase) {
        const _Float16* W0  = w6 + (phase ? 3 : 0) * MATN;   // Whh0
        const _Float16* W1i = w6 + (phase ? 4 : 1) * MATN;   // Wih1
        const _Float16* W1h = w6 + (phase ? 5 : 2) * MATN;   // Whh1
        const float* b0   = biases + (phase ? 2048 : 0);
        const float* b1   = biases + (phase ? 3072 : 1024);
        const float* wih0 = phase ? dWih0 : eWih0;

        for (int t = 0; t < 512; ++t) {
            // ================= layer 0: gates = h0_prev @ Whh0^T + b0 =================
            {
                v8f acc[4];
                float bv = b0[ncol];
                #pragma unroll
                for (int mt = 0; mt < 4; ++mt) {
                    #pragma unroll
                    for (int r = 0; r < 8; ++r) acc[mt][r] = bv;
                }
                gemm_slab(h0buf + p * 32768, W0, m0, arow, kb, ncol, acc);
                #pragma unroll
                for (int mt = 0; mt < 4; ++mt) {
                    #pragma unroll
                    for (int r = 0; r < 8; ++r)
                        s_gate[gate * 2048 + (mt * 16 + r + 8 * (ln >> 4)) * 32 + jl0 + (ln & 15)] = acc[mt][r];
                }
            }
            __syncthreads();
            {   // elementwise: add rank-1 input term, activate, update c0, emit h0_new (f16)
                _Float16* hdst = h0buf + (p ^ 1) * 32768;
                #pragma unroll
                for (int q = 0; q < 8; ++q) {
                    int idx = q * 256 + tid;
                    int bl = idx >> 5, jl = idx & 31;
                    int b = m0 + bl, j = cg * 32 + jl;
                    float xin = phase ? dec_in[b] : x[b * 512 + t];
                    float gi = s_gate[0 * 2048 + bl * 32 + jl] + xin * wih0[j];
                    float gf = s_gate[1 * 2048 + bl * 32 + jl] + xin * wih0[256 + j];
                    float gg = s_gate[2 * 2048 + bl * 32 + jl] + xin * wih0[512 + j];
                    float go = s_gate[3 * 2048 + bl * 32 + jl] + xin * wih0[768 + j];
                    float iv = sigf(gi), fv = sigf(gf), gv = tanhfast(gg), ov = sigf(go);
                    c0[q] = fv * c0[q] + iv * gv;
                    hdst[b * 256 + j] = (_Float16)(ov * tanhfast(c0[q]));
                }
            }
            grid_barrier(bcnt, bgen);   // publish h0_new

            // ========= layer 1: gates = h0_new @ Wih1^T + h1_prev @ Whh1^T + b1 =========
            {
                v8f acc[4];
                float bv = b1[ncol];
                #pragma unroll
                for (int mt = 0; mt < 4; ++mt) {
                    #pragma unroll
                    for (int r = 0; r < 8; ++r) acc[mt][r] = bv;
                }
                gemm_slab(h0buf + (p ^ 1) * 32768, W1i, m0, arow, kb, ncol, acc);
                gemm_slab(h1buf + p       * 32768, W1h, m0, arow, kb, ncol, acc);
                #pragma unroll
                for (int mt = 0; mt < 4; ++mt) {
                    #pragma unroll
                    for (int r = 0; r < 8; ++r)
                        s_gate[gate * 2048 + (mt * 16 + r + 8 * (ln >> 4)) * 32 + jl0 + (ln & 15)] = acc[mt][r];
                }
            }
            __syncthreads();
            {
                _Float16* hdst = h1buf + (p ^ 1) * 32768;
                #pragma unroll
                for (int q = 0; q < 8; ++q) {
                    int idx = q * 256 + tid;
                    int bl = idx >> 5, jl = idx & 31;
                    int b = m0 + bl, j = cg * 32 + jl;
                    float gi = s_gate[0 * 2048 + bl * 32 + jl];
                    float gf = s_gate[1 * 2048 + bl * 32 + jl];
                    float gg = s_gate[2 * 2048 + bl * 32 + jl];
                    float go = s_gate[3 * 2048 + bl * 32 + jl];
                    float iv = sigf(gi), fv = sigf(gf), gv = tanhfast(gg), ov = sigf(go);
                    c1[q] = fv * c1[q] + iv * gv;
                    hdst[b * 256 + j] = (_Float16)(ov * tanhfast(c1[q]));
                }
            }
            grid_barrier(bcnt, bgen);   // publish h1_new

            if (phase) {
                // output projection out = h1_new @ outW^T + outB, feeds next decoder input
                if (cg == 0 && tid < 64) {
                    int b = m0 + tid;
                    const _Float16* hr = h1buf + (p ^ 1) * 32768 + b * 256;
                    float acc = outB[0];
                    #pragma unroll 8
                    for (int jj = 0; jj < 256; ++jj) acc += (float)hr[jj] * outW[jj];
                    out[b * 512 + t] = acc;
                    dec_in[b] = acc;
                }
                grid_barrier(bcnt, bgen);   // publish dec_in
            }
            p ^= 1;
        }
    }
}

extern "C" void kernel_launch(void* const* d_in, const int* in_sizes, int n_in,
                              void* d_out, int out_size, void* d_ws, size_t ws_size,
                              hipStream_t stream) {
    (void)in_sizes; (void)n_in; (void)out_size; (void)ws_size;

    const float* x     = (const float*)d_in[0];
    const float* eWih0 = (const float*)d_in[1];
    const float* eWhh0 = (const float*)d_in[2];
    const float* eBih0 = (const float*)d_in[3];
    const float* eBhh0 = (const float*)d_in[4];
    const float* eWih1 = (const float*)d_in[5];
    const float* eWhh1 = (const float*)d_in[6];
    const float* eBih1 = (const float*)d_in[7];
    const float* eBhh1 = (const float*)d_in[8];
    const float* dWih0 = (const float*)d_in[9];
    const float* dWhh0 = (const float*)d_in[10];
    const float* dBih0 = (const float*)d_in[11];
    const float* dBhh0 = (const float*)d_in[12];
    const float* dWih1 = (const float*)d_in[13];
    const float* dWhh1 = (const float*)d_in[14];
    const float* dBih1 = (const float*)d_in[15];
    const float* dBhh1 = (const float*)d_in[16];
    const float* outW  = (const float*)d_in[17];
    const float* outB  = (const float*)d_in[18];

    // workspace layout (all offsets 32B-aligned)
    char* ws = (char*)d_ws;
    _Float16* w6     = (_Float16*)(ws);                 // 6 * 262144 f16 = 3 MiB
    float*    biases = (float*)   (ws + 3145728);       // eb0,eb1,db0,db1 (4*1024 f32)
    _Float16* h0     = (_Float16*)(ws + 3162112);       // double-buffered h0: 2*32768 f16
    _Float16* h1     = (_Float16*)(ws + 3293184);       // double-buffered h1: 2*32768 f16
    float*    dec_in = (float*)   (ws + 3424256);       // 128 f32
    unsigned* bar    = (unsigned*)(ws + 3424768);       // {count, generation}

    // convert the 6 recurrent/input GEMM weight matrices to f16
    k_conv6<<<6144, 256, 0, stream>>>(eWhh0, eWih1, eWhh1, dWhh0, dWih1, dWhh1, w6);
    // fuse biases, zero h state, seed decoder input with x[:,0], reset barrier
    k_misc<<<256, 256, 0, stream>>>(eBih0, eBhh0, eBih1, eBhh1, dBih0, dBhh0, dBih1, dBhh1,
                                    x, biases, h0, h1, dec_in, bar);
    // persistent recurrent kernel: 16 WGs x 8 wave32
    lstm_seq2seq<<<NWG, 256, 0, stream>>>(x, eWih0, dWih0, outW, outB,
                                          w6, biases, h0, h1, dec_in, bar,
                                          (float*)d_out);
}
